// DGCNN_71579924955362
// MI455X (gfx1250) — matrix operand order repaired and checked by reading, not compiled, not run on hardware
//
#include <hip/hip_runtime.h>

#define NODES 100000
#define KNBR  16
#define EDGES (NODES * KNBR)

typedef __attribute__((ext_vector_type(16))) __bf16 v16bf;
typedef __attribute__((ext_vector_type(8)))  __bf16 v8bf;
typedef __attribute__((ext_vector_type(8)))  float  v8f;
typedef __attribute__((ext_vector_type(4)))  float  v4f;

union V16U { v16bf v; v8bf h[2]; };

static __device__ __forceinline__ v8f wmma_bf16(v16bf a, v16bf b, v8f c) {
  // (neg_a, A, neg_b, B, c_mod, C, reuse_a, reuse_b)
  return __builtin_amdgcn_wmma_f32_16x16x32_bf16(false, a, false, b, (short)0, c,
                                                 false, false);
}

// ---------------------------------------------------------------------------
// Pack f32 weights [Kdim x Nsrc] (row-major) into bf16 WMMA B-operand layout:
// per (kt,nt) tile of 512 bf16: lane L holds column n = nt*16 + (L&15),
// element j -> k = kt*32 + (L>>4)*16 + j.  One contiguous 32B frag per lane.
// ---------------------------------------------------------------------------
__global__ void pack_weights(const float* __restrict__ src, __bf16* __restrict__ dst,
                             int Nsrc, int ntiles, int total) {
  int idx = blockIdx.x * blockDim.x + threadIdx.x;
  if (idx >= total) return;
  int tile   = idx >> 9;
  int within = idx & 511;
  int lane   = within >> 4;
  int j      = within & 15;
  int half   = lane >> 4;
  int kt = tile / ntiles;
  int nt = tile - kt * ntiles;
  int k = kt * 32 + half * 16 + j;
  int n = nt * 16 + (lane & 15);
  float v = (n < Nsrc) ? src[k * Nsrc + n] : 0.0f;
  dst[idx] = (__bf16)v;
}

// A fragment from a row-major bf16 LDS tile (stride bf16 elems per row).
// 16-bit A 16x32 layout (matches packed B): lane(m=L&15, half=L>>4):
//   j=0..15 -> k = ktBase + half*16 + j
static __device__ __forceinline__ v16bf load_afrag(const __bf16* tile, int stride,
                                                   int mrow, int half, int ktBase) {
  V16U u;
  const __bf16* base = tile + mrow * stride + ktBase + half * 16;
  u.h[0] = *(const v8bf*)(base);
  u.h[1] = *(const v8bf*)(base + 8);
  return u.v;
}

// ---------------------------------------------------------------------------
// Kernel 1: conv1 for the FIRST N edges only (only h1[0:N] is ever read).
// One wave per 16-edge tile. h1 out: [N,64] f32.
// ---------------------------------------------------------------------------
__global__ __launch_bounds__(256) void conv1_kernel(
    const float* __restrict__ x, const int* __restrict__ colIdx,
    const __bf16* __restrict__ pW1, const float* __restrict__ b1,
    const __bf16* __restrict__ pW2, const float* __restrict__ b2,
    float* __restrict__ h1) {
  __shared__ __bf16 lds[8 * 2048];
  const int lane = threadIdx.x & 31;
  const int wave = threadIdx.x >> 5;
  const int half = lane >> 4;
  const int mrow = lane & 15;
  __bf16* tile = lds + wave * 2048;

  int t = blockIdx.x * 8 + wave;
  if (t >= NODES / 16) return;

  // Build A tile 16x32: row k = edge t*16+k, feats [x[r] | x[c]-x[r]]
  int e = t * 16 + mrow;
  int r = e >> 4;                 // row = repeat(arange(N), 16)
  int c = colIdx[e];
  const v4f* xr = (const v4f*)(x + r * 16);
  const v4f* xc = (const v4f*)(x + c * 16);
#pragma unroll
  for (int q = 0; q < 4; ++q) {
    v4f a = xr[q];
    v4f v = a;
    if (half) { v4f b = xc[q]; v.x = b.x - a.x; v.y = b.y - a.y; v.z = b.z - a.z; v.w = b.w - a.w; }
    __bf16* p = tile + mrow * 32 + half * 16 + q * 4;
    p[0] = (__bf16)v.x; p[1] = (__bf16)v.y; p[2] = (__bf16)v.z; p[3] = (__bf16)v.w;
  }

  // GEMM1: [16x32] @ W1[32x64]  (bias preloaded into accumulator)
  v16bf af = load_afrag(tile, 32, mrow, half, 0);
  v8f hacc[4];
#pragma unroll
  for (int nt = 0; nt < 4; ++nt) {
    float bias = b1[nt * 16 + mrow];
    v8f acc;
#pragma unroll
    for (int i = 0; i < 8; ++i) acc[i] = bias;
    v16bf bf = *(const v16bf*)(pW1 + nt * 512 + lane * 16);
    hacc[nt] = wmma_bf16(af, bf, acc);
  }

  // relu -> H (16x64 bf16) at tile+512
  __bf16* H = tile + 512;
#pragma unroll
  for (int nt = 0; nt < 4; ++nt)
#pragma unroll
    for (int rr = 0; rr < 8; ++rr) {
      float v = hacc[nt][rr]; v = v > 0.f ? v : 0.f;
      int m = rr + half * 8;
      H[m * 64 + nt * 16 + mrow] = (__bf16)v;
    }

  // GEMM2: [16x64] @ W2[64x64]
  v8f acc2[4];
#pragma unroll
  for (int nt = 0; nt < 4; ++nt) {
    float bias = b2[nt * 16 + mrow];
#pragma unroll
    for (int i = 0; i < 8; ++i) acc2[nt][i] = bias;
  }
#pragma unroll
  for (int kt = 0; kt < 2; ++kt) {
    v16bf a2 = load_afrag(H, 64, mrow, half, kt * 32);
#pragma unroll
    for (int nt = 0; nt < 4; ++nt) {
      v16bf bf = *(const v16bf*)(pW2 + (kt * 4 + nt) * 512 + lane * 16);
      acc2[nt] = wmma_bf16(a2, bf, acc2[nt]);
    }
  }

  // relu + store h1 f32 (coalesced: lanes 0-15 / 16-31 each cover one row)
#pragma unroll
  for (int nt = 0; nt < 4; ++nt)
#pragma unroll
    for (int rr = 0; rr < 8; ++rr) {
      float v = acc2[nt][rr]; v = v > 0.f ? v : 0.f;
      int m = rr + half * 8;
      h1[(t * 16 + m) * 64 + nt * 16 + mrow] = v;
    }
}

// ---------------------------------------------------------------------------
// Kernel 2: conv2 + max over K.  One wave per TWO nodes: each node's 16 edges
// form one M=16 tile; both tiles share every B fragment (halves weight
// traffic per WMMA).  g out: [N,128] f32.
// ---------------------------------------------------------------------------
__global__ __launch_bounds__(256) void conv2_kernel(
    const float* __restrict__ h1, const int* __restrict__ colIdx,
    const __bf16* __restrict__ pW3, const float* __restrict__ b3,
    const __bf16* __restrict__ pW4, const float* __restrict__ b4,
    float* __restrict__ g) {
  __shared__ __bf16 lds[8 * 4096];   // 2 tiles of 16x128 bf16 per wave
  const int lane = threadIdx.x & 31;
  const int wave = threadIdx.x >> 5;
  const int half = lane >> 4;
  const int mrow = lane & 15;
  __bf16* tile = lds + wave * 4096;

  int base = (blockIdx.x * 8 + wave) * 2;
  if (base >= NODES) return;

  // Build both A tiles 16x128: row k -> [h1[node] | h1[col_k]-h1[node]]
#pragma unroll
  for (int p = 0; p < 2; ++p) {
    int node = base + p;
    int c = colIdx[node * 16 + mrow];
    const v4f* hn = (const v4f*)(h1 + node * 64);
    const v4f* hc = (const v4f*)(h1 + c * 64);
    __bf16* tp = tile + p * 2048;
#pragma unroll
    for (int q = 0; q < 16; ++q) {
      v4f a = hn[q];
      v4f v = a;
      if (half) { v4f b = hc[q]; v.x = b.x - a.x; v.y = b.y - a.y; v.z = b.z - a.z; v.w = b.w - a.w; }
      __bf16* pdst = tp + mrow * 128 + half * 64 + q * 4;
      pdst[0] = (__bf16)v.x; pdst[1] = (__bf16)v.y; pdst[2] = (__bf16)v.z; pdst[3] = (__bf16)v.w;
    }
  }

  // GEMM1: [2 x 16x128] @ W3[128x128]
  v8f acc[2][8];
#pragma unroll
  for (int nt = 0; nt < 8; ++nt) {
    float bias = b3[nt * 16 + mrow];
#pragma unroll
    for (int i = 0; i < 8; ++i) { acc[0][nt][i] = bias; acc[1][nt][i] = bias; }
  }
#pragma unroll
  for (int kt = 0; kt < 4; ++kt) {
    v16bf af0 = load_afrag(tile,        128, mrow, half, kt * 32);
    v16bf af1 = load_afrag(tile + 2048, 128, mrow, half, kt * 32);
#pragma unroll
    for (int nt = 0; nt < 8; ++nt) {
      v16bf bf = *(const v16bf*)(pW3 + (kt * 8 + nt) * 512 + lane * 16);
      acc[0][nt] = wmma_bf16(af0, bf, acc[0][nt]);
      acc[1][nt] = wmma_bf16(af1, bf, acc[1][nt]);
    }
  }

  // relu -> overwrite tiles with H (A fully consumed)
#pragma unroll
  for (int p = 0; p < 2; ++p)
#pragma unroll
    for (int nt = 0; nt < 8; ++nt)
#pragma unroll
      for (int rr = 0; rr < 8; ++rr) {
        float v = acc[p][nt][rr]; v = v > 0.f ? v : 0.f;
        int m = rr + half * 8;
        tile[p * 2048 + m * 128 + nt * 16 + mrow] = (__bf16)v;
      }

  // GEMM2: [2 x 16x128] @ W4[128x128]
  v8f acc2[2][8];
#pragma unroll
  for (int nt = 0; nt < 8; ++nt) {
    float bias = b4[nt * 16 + mrow];
#pragma unroll
    for (int i = 0; i < 8; ++i) { acc2[0][nt][i] = bias; acc2[1][nt][i] = bias; }
  }
#pragma unroll
  for (int kt = 0; kt < 4; ++kt) {
    v16bf af0 = load_afrag(tile,        128, mrow, half, kt * 32);
    v16bf af1 = load_afrag(tile + 2048, 128, mrow, half, kt * 32);
#pragma unroll
    for (int nt = 0; nt < 8; ++nt) {
      v16bf bf = *(const v16bf*)(pW4 + (kt * 8 + nt) * 512 + lane * 16);
      acc2[0][nt] = wmma_bf16(af0, bf, acc2[0][nt]);
      acc2[1][nt] = wmma_bf16(af1, bf, acc2[1][nt]);
    }
  }

  // relu + max over the 16 edge rows (8 acc rows + cross-half shuffle) -> g
#pragma unroll
  for (int p = 0; p < 2; ++p) {
    int node = base + p;
#pragma unroll
    for (int nt = 0; nt < 8; ++nt) {
      float m = 0.0f;  // relu floor
#pragma unroll
      for (int rr = 0; rr < 8; ++rr) m = fmaxf(m, acc2[p][nt][rr]);
      m = fmaxf(m, __shfl_xor(m, 16, 32));
      if (half == 0) g[node * 128 + nt * 16 + mrow] = m;
    }
  }
}

// ---------------------------------------------------------------------------
// Kernel 3: out = relu(g@W5+b5)@W6+b6 over 16-node tiles. out: [N,40] f32.
// ---------------------------------------------------------------------------
__global__ __launch_bounds__(256) void head_kernel(
    const float* __restrict__ g,
    const __bf16* __restrict__ pW5, const float* __restrict__ b5,
    const __bf16* __restrict__ pW6, const float* __restrict__ b6,
    float* __restrict__ out) {
  __shared__ __bf16 lds[8 * 2048];
  const int lane = threadIdx.x & 31;
  const int wave = threadIdx.x >> 5;
  const int half = lane >> 4;
  const int mrow = lane & 15;
  __bf16* tile = lds + wave * 2048;

  int t = blockIdx.x * 8 + wave;
  if (t >= NODES / 16) return;

  // Build A tile 16x128 = g rows (bf16)
  const v4f* gr = (const v4f*)(g + (t * 16 + mrow) * 128 + half * 64);
#pragma unroll
  for (int q = 0; q < 16; ++q) {
    v4f v = gr[q];
    __bf16* p = tile + mrow * 128 + half * 64 + q * 4;
    p[0] = (__bf16)v.x; p[1] = (__bf16)v.y; p[2] = (__bf16)v.z; p[3] = (__bf16)v.w;
  }

  // GEMM1: [16x128] @ W5[128x128]
  v8f acc[8];
#pragma unroll
  for (int nt = 0; nt < 8; ++nt) {
    float bias = b5[nt * 16 + mrow];
#pragma unroll
    for (int i = 0; i < 8; ++i) acc[nt][i] = bias;
  }
#pragma unroll
  for (int kt = 0; kt < 4; ++kt) {
    v16bf af = load_afrag(tile, 128, mrow, half, kt * 32);
#pragma unroll
    for (int nt = 0; nt < 8; ++nt) {
      v16bf bf = *(const v16bf*)(pW5 + (kt * 8 + nt) * 512 + lane * 16);
      acc[nt] = wmma_bf16(af, bf, acc[nt]);
    }
  }

  // relu -> H (overwrite tile)
#pragma unroll
  for (int nt = 0; nt < 8; ++nt)
#pragma unroll
    for (int rr = 0; rr < 8; ++rr) {
      float v = acc[nt][rr]; v = v > 0.f ? v : 0.f;
      int m = rr + half * 8;
      tile[m * 128 + nt * 16 + mrow] = (__bf16)v;
    }

  // GEMM2: [16x128] @ W6[128x48] (cols 40..47 are zero padding), NO final relu
  int col2 = 32 + mrow;
  v8f acc2[3];
  float bb[3];
  bb[0] = b6[mrow]; bb[1] = b6[16 + mrow]; bb[2] = (col2 < 40) ? b6[col2] : 0.0f;
#pragma unroll
  for (int nt = 0; nt < 3; ++nt)
#pragma unroll
    for (int i = 0; i < 8; ++i) acc2[nt][i] = bb[nt];
#pragma unroll
  for (int kt = 0; kt < 4; ++kt) {
    v16bf af = load_afrag(tile, 128, mrow, half, kt * 32);
#pragma unroll
    for (int nt = 0; nt < 3; ++nt) {
      v16bf bf = *(const v16bf*)(pW6 + (kt * 3 + nt) * 512 + lane * 16);
      acc2[nt] = wmma_bf16(af, bf, acc2[nt]);
    }
  }

  // store out[node][col] for col < 40
#pragma unroll
  for (int nt = 0; nt < 3; ++nt) {
    int col = nt * 16 + mrow;
    if (col < 40) {
#pragma unroll
      for (int rr = 0; rr < 8; ++rr) {
        int m = rr + half * 8;
        out[(t * 16 + m) * 40 + col] = acc2[nt][rr];
      }
    }
  }
}

// ---------------------------------------------------------------------------
// Workspace layout (bytes):
//   pW1 @ 0      (4096)   pW2 @ 4096   (8192)   pW3 @ 12288 (32768)
//   pW4 @ 45056  (32768)  pW5 @ 77824  (32768)  pW6 @ 110592 (12288)
//   h1  @ 131072 (N*64*4 = 25,600,000)
//   g   @ 131072 + 25,600,000 (N*128*4 = 51,200,000)
// ---------------------------------------------------------------------------
extern "C" void kernel_launch(void* const* d_in, const int* in_sizes, int n_in,
                              void* d_out, int out_size, void* d_ws, size_t ws_size,
                              hipStream_t stream) {
  const float* x  = (const float*)d_in[0];
  const int*   ei = (const int*)d_in[1];
  const float* W1 = (const float*)d_in[2];  const float* b1 = (const float*)d_in[3];
  const float* W2 = (const float*)d_in[4];  const float* b2 = (const float*)d_in[5];
  const float* W3 = (const float*)d_in[6];  const float* b3 = (const float*)d_in[7];
  const float* W4 = (const float*)d_in[8];  const float* b4 = (const float*)d_in[9];
  const float* W5 = (const float*)d_in[10]; const float* b5 = (const float*)d_in[11];
  const float* W6 = (const float*)d_in[12]; const float* b6 = (const float*)d_in[13];

  const int* colIdx = ei + EDGES;  // edge_index[1] (second row)

  char* ws = (char*)d_ws;
  __bf16* pW1 = (__bf16*)(ws + 0);
  __bf16* pW2 = (__bf16*)(ws + 4096);
  __bf16* pW3 = (__bf16*)(ws + 12288);
  __bf16* pW4 = (__bf16*)(ws + 45056);
  __bf16* pW5 = (__bf16*)(ws + 77824);
  __bf16* pW6 = (__bf16*)(ws + 110592);
  float*  h1  = (float*)(ws + 131072);
  float*  gbuf = (float*)(ws + 131072 + (size_t)NODES * 64 * 4);
  float*  out = (float*)d_out;

  // Pack weights to WMMA B layout (bf16)
  pack_weights<<<8,  256, 0, stream>>>(W1, pW1, 64, 4, 2048);    // 32x64:  1x4 tiles
  pack_weights<<<16, 256, 0, stream>>>(W2, pW2, 64, 4, 4096);    // 64x64:  2x4
  pack_weights<<<64, 256, 0, stream>>>(W3, pW3, 128, 8, 16384);  // 128x128: 4x8
  pack_weights<<<64, 256, 0, stream>>>(W4, pW4, 128, 8, 16384);  // 128x128: 4x8
  pack_weights<<<64, 256, 0, stream>>>(W5, pW5, 128, 8, 16384);  // 128x128: 4x8
  pack_weights<<<24, 256, 0, stream>>>(W6, pW6, 40, 3, 6144);    // 128x48: 4x3 (padded)

  // conv1 over first N edges (6250 tiles of 16 edges, 8 waves/block)
  conv1_kernel<<<(NODES / 16 + 7) / 8, 256, 0, stream>>>(x, colIdx, pW1, b1, pW2, b2, h1);
  // conv2 + max: one wave per 2 nodes (2 M-tiles share each B fragment)
  conv2_kernel<<<NODES / 16, 256, 0, stream>>>(h1, colIdx, pW3, b3, pW4, b4, gbuf);
  // head MLP over 16-node tiles
  head_kernel<<<(NODES / 16 + 7) / 8, 256, 0, stream>>>(gbuf, pW5, b5, pW6, b6, out);
}
